// NmtFlax_51273319580297
// MI455X (gfx1250) — matrix operand order repaired and checked by reading, not compiled, not run on hardware
//
#include <hip/hip_runtime.h>
#include <hip/hip_bf16.h>

#define B_ 32
#define S_ 48
#define T_ 48
#define E_ 256
#define H_ 256
#define V_ 32000
#define G4 1024 /* 4*H */

typedef __attribute__((ext_vector_type(16))) __bf16 v16bf;
typedef __attribute__((ext_vector_type(8)))  float  v8f;

union FragBF { uint4 q[2]; v16bf v; };

__device__ __forceinline__ unsigned short f2bf(float f) {
  union { float f; unsigned u; } c; c.f = f;
  unsigned r = c.u + 0x7FFFu + ((c.u >> 16) & 1u);
  return (unsigned short)(r >> 16);
}

__device__ __forceinline__ float sigm(float x) { return 1.0f / (1.0f + __expf(-x)); }

// ---------------------------------------------------------------------------
// WMMA K-loop, one 16x16 f32 tile. A row-major (M x K) bf16, Bt row-major
// (N x K) bf16 (B pre-transposed). Pointers already offset to tile start.
// A-fragment lane layout per CDNA5 ISA (16-bit A 16x32):
//   lanes 0-15  : row = lane,    K = {0..7, 16..23}
//   lanes 16-31 : row = lane-16, K = {8..15, 24..31}
// B-fragment: lane holds one column, same K order as A.
// ---------------------------------------------------------------------------
__device__ __forceinline__ v8f wmma_kloop(const unsigned short* __restrict__ A, int lda,
                                          const unsigned short* __restrict__ Bt, int ldb,
                                          int K, v8f acc) {
  const int lane = threadIdx.x & 31;
  const int half = lane >> 4;
  const int l15  = lane & 15;
  const unsigned short* arow = A + (long)l15 * lda;
  const unsigned short* brow = Bt + (long)l15 * ldb;
  for (int k0 = 0; k0 < K; k0 += 32) {
    FragBF a, b;
    const int ka = k0 + half * 8;
    a.q[0] = *reinterpret_cast<const uint4*>(arow + ka);
    a.q[1] = *reinterpret_cast<const uint4*>(arow + ka + 16);
    const int kb = k0 + half * 8;
    b.q[0] = *reinterpret_cast<const uint4*>(brow + kb);
    b.q[1] = *reinterpret_cast<const uint4*>(brow + kb + 16);
    acc = __builtin_amdgcn_wmma_f32_16x16x32_bf16(false, a.v, false, b.v,
                                                  (short)0, acc, false, false);
  }
  return acc;
}

// Dual-M K-loop for M=32 GEMMs: one B fragment feeds two accumulators
// (rows 0..15 and 16..31), halving B-side load traffic.
__device__ __forceinline__ void wmma_kloop2(const unsigned short* __restrict__ A, int lda,
                                            const unsigned short* __restrict__ Bt, int ldb,
                                            int K, v8f& acc0, v8f& acc1) {
  const int lane = threadIdx.x & 31;
  const int half = lane >> 4;
  const int l15  = lane & 15;
  const unsigned short* arow0 = A + (long)l15 * lda;
  const unsigned short* arow1 = A + (long)(16 + l15) * lda;
  const unsigned short* brow  = Bt + (long)l15 * ldb;
  for (int k0 = 0; k0 < K; k0 += 32) {
    FragBF a0, a1, b;
    const int ka = k0 + half * 8;
    a0.q[0] = *reinterpret_cast<const uint4*>(arow0 + ka);
    a0.q[1] = *reinterpret_cast<const uint4*>(arow0 + ka + 16);
    a1.q[0] = *reinterpret_cast<const uint4*>(arow1 + ka);
    a1.q[1] = *reinterpret_cast<const uint4*>(arow1 + ka + 16);
    const int kb = k0 + half * 8;
    b.q[0] = *reinterpret_cast<const uint4*>(brow + kb);
    b.q[1] = *reinterpret_cast<const uint4*>(brow + kb + 16);
    acc0 = __builtin_amdgcn_wmma_f32_16x16x32_bf16(false, a0.v, false, b.v,
                                                   (short)0, acc0, false, false);
    acc1 = __builtin_amdgcn_wmma_f32_16x16x32_bf16(false, a1.v, false, b.v,
                                                   (short)0, acc1, false, false);
  }
}

// ---------------- batched (tall-M) GEMM: C = A @ Bt^T + bias --------------
__global__ void k_gemm_bias(const unsigned short* __restrict__ A, int lda,
                            const unsigned short* __restrict__ Bt, int ldb,
                            const float* __restrict__ bias,
                            float* __restrict__ C, long ldc,
                            int Mt, int ntiles, int K) {
  int tile = blockIdx.x * 4 + (threadIdx.x >> 5);
  if (tile >= ntiles) return;
  int m0 = (tile % Mt) * 16;
  int n0 = (tile / Mt) * 16;
  v8f acc = {};
  acc = wmma_kloop(A + (long)m0 * lda, lda, Bt + (long)n0 * ldb, ldb, K, acc);
  const int lane = threadIdx.x & 31;
  const int half = lane >> 4;
  const int col  = n0 + (lane & 15);
  float bv = bias ? bias[col] : 0.0f;
#pragma unroll
  for (int r = 0; r < 8; ++r) {
    int row = m0 + r + half * 8;
    C[(long)row * ldc + col] = acc[r] + bv;
  }
}

// ---------------- M=32 GEMMs (dual accumulator per wave) ------------------
// C(32 x N) = A(32 x K) @ Bt^T + bias
__global__ void k_gemm_bias_m32(const unsigned short* __restrict__ A, int lda,
                                const unsigned short* __restrict__ Bt, int ldb,
                                const float* __restrict__ bias,
                                float* __restrict__ C, long ldc,
                                int ntilesN, int K) {
  int tile = blockIdx.x * 4 + (threadIdx.x >> 5);
  if (tile >= ntilesN) return;
  int n0 = tile * 16;
  const unsigned short* bt = Bt + (long)n0 * ldb;
  __builtin_prefetch(bt, 0, 1);                     // global_prefetch_b8
  v8f acc0 = {}, acc1 = {};
  wmma_kloop2(A, lda, bt, ldb, K, acc0, acc1);
  const int lane = threadIdx.x & 31;
  const int half = lane >> 4;
  const int col  = n0 + (lane & 15);
  float bv = bias ? bias[col] : 0.0f;
#pragma unroll
  for (int r = 0; r < 8; ++r) {
    int row = r + half * 8;
    C[(long)row * ldc + col]        = acc0[r] + bv;
    C[(long)(row + 16) * ldc + col] = acc1[r] + bv;
  }
}

// C(32 x N) = A(32 x K) @ Bt^T + Add   (Add = precomputed x@Wx+b slice)
__global__ void k_gemm_add_m32(const unsigned short* __restrict__ A, int lda,
                               const unsigned short* __restrict__ Bt, int ldb,
                               const float* __restrict__ Add, int ldadd,
                               float* __restrict__ C, int ldc,
                               int ntilesN, int K) {
  int tile = blockIdx.x * 4 + (threadIdx.x >> 5);
  if (tile >= ntilesN) return;
  int n0 = tile * 16;
  v8f acc0 = {}, acc1 = {};
  wmma_kloop2(A, lda, Bt + (long)n0 * ldb, ldb, K, acc0, acc1);
  const int lane = threadIdx.x & 31;
  const int half = lane >> 4;
  const int col  = n0 + (lane & 15);
#pragma unroll
  for (int r = 0; r < 8; ++r) {
    int row = r + half * 8;
    C[row * ldc + col]        = acc0[r] + Add[row * ldadd + col];
    C[(row + 16) * ldc + col] = acc1[r] + Add[(row + 16) * ldadd + col];
  }
}

// O1/O2 (bf16) = tanh(A(32 x K) @ Bt^T + bias)   (combine layer)
__global__ void k_gemm_tanh2_m32(const unsigned short* __restrict__ A, int lda,
                                 const unsigned short* __restrict__ Bt, int ldb,
                                 const float* __restrict__ bias,
                                 unsigned short* __restrict__ O1, int ld1,
                                 unsigned short* __restrict__ O2, int ld2,
                                 int ntilesN, int K) {
  int tile = blockIdx.x * 4 + (threadIdx.x >> 5);
  if (tile >= ntilesN) return;
  int n0 = tile * 16;
  v8f acc0 = {}, acc1 = {};
  wmma_kloop2(A, lda, Bt + (long)n0 * ldb, ldb, K, acc0, acc1);
  const int lane = threadIdx.x & 31;
  const int half = lane >> 4;
  const int col  = n0 + (lane & 15);
  float bv = bias[col];
#pragma unroll
  for (int r = 0; r < 8; ++r) {
    int row = r + half * 8;
    unsigned short o0 = f2bf(tanhf(acc0[r] + bv));
    unsigned short o1 = f2bf(tanhf(acc1[r] + bv));
    O1[row * ld1 + col] = o0;
    O2[row * ld2 + col] = o0;
    O1[(row + 16) * ld1 + col] = o1;
    O2[(row + 16) * ld2 + col] = o1;
  }
}

// out[n*ldout + k] = bf16(in[k*ldin + n])   (weight transpose to N-major bf16)
__global__ void k_transpose_bf16(const float* __restrict__ in, int ldin,
                                 unsigned short* __restrict__ out, int ldout,
                                 int K, int N) {
  int idx = blockIdx.x * blockDim.x + threadIdx.x;
  if (idx >= K * N) return;
  int n = idx / K, k = idx % K;
  out[(long)n * ldout + k] = f2bf(in[(long)k * ldin + n]);
}

// x = conv1d_SAME_k2(src_embed[src_ids]) + conv_b ; also bf16 + reversed bf16
__global__ void k_embed_conv(const int* __restrict__ ids, const float* __restrict__ emb,
                             const float* __restrict__ ck, const float* __restrict__ cb,
                             float* __restrict__ x, unsigned short* __restrict__ xb,
                             unsigned short* __restrict__ xrb) {
  int idx = blockIdx.x * blockDim.x + threadIdx.x;
  if (idx >= B_ * S_ * E_) return;
  int e = idx % E_;
  int s = (idx / E_) % S_;
  int b = idx / (E_ * S_);
  const float* e0 = emb + (long)ids[b * S_ + s] * E_;
  float acc = cb[e];
  for (int i = 0; i < E_; ++i) acc += e0[i] * ck[i * E_ + e];
  if (s + 1 < S_) {  // SAME padding for k=2: pad right only
    const float* e1 = emb + (long)ids[b * S_ + s + 1] * E_;
    const float* k1 = ck + E_ * E_;
    for (int i = 0; i < E_; ++i) acc += e1[i] * k1[i * E_ + e];
  }
  x[idx] = acc;
  unsigned short bf = f2bf(acc);
  xb[idx] = bf;
  xrb[(b * S_ + (S_ - 1 - s)) * E_ + e] = bf;
}

__global__ void k_embed_tgt(const int* __restrict__ ids, const float* __restrict__ emb,
                            unsigned short* __restrict__ yb) {
  int idx = blockIdx.x * blockDim.x + threadIdx.x;
  if (idx >= B_ * T_ * E_) return;
  int e = idx % E_;
  int t = (idx / E_) % T_;
  int b = idx / (E_ * T_);
  yb[idx] = f2bf(emb[(long)ids[b * T_ + t] * E_ + e]);
}

// Encoder pointwise LSTM step for both directions (d=0 fwd, d=1 bwd).
__global__ void k_enc_point(const float* __restrict__ gf, const float* __restrict__ gb,
                            float* __restrict__ cf, float* __restrict__ hf,
                            unsigned short* __restrict__ hfb,
                            float* __restrict__ cbk, float* __restrict__ hb,
                            unsigned short* __restrict__ hbb,
                            float* __restrict__ hs_f, float* __restrict__ hs_b,
                            const int* __restrict__ lens, int s) {
  int idx = blockIdx.x * blockDim.x + threadIdx.x;
  if (idx >= 2 * B_ * H_) return;
  int j = idx % H_;
  int b = (idx / H_) % B_;
  int d = idx / (B_ * H_);
  const float* g = (d ? gb : gf) + b * G4;
  float ii = sigm(g[j]);
  float ff = sigm(g[H_ + j]);
  float gg = tanhf(g[2 * H_ + j]);
  float oo = sigm(g[3 * H_ + j]);
  float* c = (d ? cbk : cf) + b * H_ + j;
  float* h = (d ? hb : hf) + b * H_ + j;
  unsigned short* hbf = (d ? hbb : hfb) + b * H_ + j;
  float nc = ff * (*c) + ii * gg;
  float nh = oo * tanhf(nc);
  bool m = s < lens[b];
  float cN = m ? nc : *c;
  float hN = m ? nh : *h;
  *c = cN; *h = hN; *hbf = f2bf(hN);
  float hsv = m ? nh : 0.0f;
  if (d == 0) hs_f[(b * S_ + s) * H_ + j] = hsv;
  else        hs_b[(b * S_ + (S_ - 1 - s)) * H_ + j] = hsv;  // store pre-flipped
}

// enc = concat(hs_f, hs_b) ; also bf16 copy for the att_proj GEMM
__global__ void k_enc_concat(const float* __restrict__ hs_f, const float* __restrict__ hs_b,
                             float* __restrict__ enc, unsigned short* __restrict__ encb) {
  int idx = blockIdx.x * blockDim.x + threadIdx.x;
  if (idx >= B_ * S_ * 2 * H_) return;
  int j = idx % (2 * H_);
  int bs = idx / (2 * H_);
  float v = (j < H_) ? hs_f[bs * H_ + j] : hs_b[bs * H_ + (j - H_)];
  enc[idx] = v;
  encb[idx] = f2bf(v);
}

// h0 = [h_f,h_b]@h_proj -> A2 h-slot (bf16); c0 = [c_f,c_b]@c_proj -> c_dec
__global__ void k_proj_hc(const float* __restrict__ hf, const float* __restrict__ hb,
                          const float* __restrict__ cf, const float* __restrict__ cbk,
                          const float* __restrict__ hp, const float* __restrict__ cp,
                          float* __restrict__ c_dec, unsigned short* __restrict__ A2) {
  int idx = blockIdx.x * blockDim.x + threadIdx.x;
  if (idx >= 2 * B_ * H_) return;
  int j = idx % H_;
  int b = (idx / H_) % B_;
  int which = idx / (B_ * H_);  // 0 -> h0, 1 -> c0
  const float* w  = which ? cp : hp;
  const float* xf = which ? cf : hf;
  const float* xb = which ? cbk : hb;
  float acc = 0.0f;
  for (int k = 0; k < H_; ++k) acc += xf[b * H_ + k] * w[k * H_ + j];
  for (int k = 0; k < H_; ++k) acc += xb[b * H_ + k] * w[(H_ + k) * H_ + j];
  if (which) c_dec[b * H_ + j] = acc;
  else       A2[b * 512 + 256 + j] = f2bf(acc);
}

// Decoder pointwise LSTM; writes dec_h (f32) + bf16 into A2 h-slot & comb_in[0:H]
__global__ void k_dec_point(const float* __restrict__ g, float* __restrict__ c_dec,
                            float* __restrict__ dech,
                            unsigned short* __restrict__ A2,
                            unsigned short* __restrict__ comb_in) {
  int idx = blockIdx.x * blockDim.x + threadIdx.x;
  if (idx >= B_ * H_) return;
  int j = idx % H_;
  int b = idx / H_;
  const float* gg = g + b * G4;
  float ii = sigm(gg[j]);
  float ff = sigm(gg[H_ + j]);
  float gt = tanhf(gg[2 * H_ + j]);
  float oo = sigm(gg[3 * H_ + j]);
  float nc = ff * c_dec[idx] + ii * gt;
  float nh = oo * tanhf(nc);
  c_dec[idx] = nc;
  dech[idx] = nh;
  unsigned short bv = f2bf(nh);
  A2[b * 512 + 256 + j] = bv;
  comb_in[b * 768 + j] = bv;
}

// Attention: scores = enc_proj . dec_h, masked softmax over S, context over enc.
// One block per batch row, 512 threads.
__global__ void k_attn(const float* __restrict__ enc_proj, const float* __restrict__ enc,
                       const float* __restrict__ dech, const int* __restrict__ lens,
                       unsigned short* __restrict__ comb_in) {
  int b = blockIdx.x;
  int t = threadIdx.x;
  __shared__ float dh[H_];
  __shared__ float al[S_];
  if (t < H_) dh[t] = dech[b * H_ + t];
  __syncthreads();
  if (t < S_) {
    float acc = 0.0f;
    const float* ep = enc_proj + (b * S_ + t) * H_;
    for (int k = 0; k < H_; ++k) acc += ep[k] * dh[k];
    al[t] = (t < lens[b]) ? acc : -1e9f;
  }
  __syncthreads();
  if (t == 0) {
    float mx = al[0];
    for (int s2 = 1; s2 < S_; ++s2) mx = fmaxf(mx, al[s2]);
    float sum = 0.0f;
    for (int s2 = 0; s2 < S_; ++s2) { float e = __expf(al[s2] - mx); al[s2] = e; sum += e; }
    float inv = 1.0f / sum;
    for (int s2 = 0; s2 < S_; ++s2) al[s2] *= inv;
  }
  __syncthreads();
  {  // context: t in [0,512)
    float acc = 0.0f;
    const float* eb = enc + b * S_ * (2 * H_) + t;
    for (int s2 = 0; s2 < S_; ++s2) acc += al[s2] * eb[s2 * (2 * H_)];
    comb_in[b * 768 + 256 + t] = f2bf(acc);
  }
}

// ---------------------------------------------------------------------------

extern "C" void kernel_launch(void* const* d_in, const int* in_sizes, int n_in,
                              void* d_out, int out_size, void* d_ws, size_t ws_size,
                              hipStream_t stream) {
  (void)in_sizes; (void)n_in; (void)out_size; (void)ws_size;
  const int*   src_ids  = (const int*)d_in[0];
  const int*   src_lens = (const int*)d_in[1];
  const int*   tgt_ids  = (const int*)d_in[2];
  const float* s_emb    = (const float*)d_in[3];
  const float* t_emb    = (const float*)d_in[4];
  const float* conv_k   = (const float*)d_in[5];
  const float* conv_b   = (const float*)d_in[6];
  const float* h_proj   = (const float*)d_in[7];
  const float* c_proj   = (const float*)d_in[8];
  const float* att_proj = (const float*)d_in[9];
  const float* efwx = (const float*)d_in[10];
  const float* efwh = (const float*)d_in[11];
  const float* efb  = (const float*)d_in[12];
  const float* ebwx = (const float*)d_in[13];
  const float* ebwh = (const float*)d_in[14];
  const float* ebb  = (const float*)d_in[15];
  const float* dwx  = (const float*)d_in[16];
  const float* dwh  = (const float*)d_in[17];
  const float* db   = (const float*)d_in[18];
  const float* cw   = (const float*)d_in[19];
  const float* cb   = (const float*)d_in[20];
  const float* vw   = (const float*)d_in[21];
  const float* vb   = (const float*)d_in[22];
  float* out = (float*)d_out;

  size_t off = 0;
  auto alloc = [&](size_t bytes) -> void* {
    void* p = (char*)d_ws + off;
    off += (bytes + 255) & ~(size_t)255;
    return p;
  };

  // bf16 N-major transposed weights
  unsigned short* vwT    = (unsigned short*)alloc((size_t)V_ * H_ * 2);
  unsigned short* efwxT  = (unsigned short*)alloc((size_t)G4 * E_ * 2);
  unsigned short* efwhT  = (unsigned short*)alloc((size_t)G4 * H_ * 2);
  unsigned short* ebwxT  = (unsigned short*)alloc((size_t)G4 * E_ * 2);
  unsigned short* ebwhT  = (unsigned short*)alloc((size_t)G4 * H_ * 2);
  unsigned short* dwxTy  = (unsigned short*)alloc((size_t)G4 * E_ * 2);       // y-part of dec_wx
  unsigned short* decohT = (unsigned short*)alloc((size_t)G4 * 512 * 2);      // [Wx_opart ; Wh]
  unsigned short* combT  = (unsigned short*)alloc((size_t)H_ * 768 * 2);
  unsigned short* attT   = (unsigned short*)alloc((size_t)H_ * 512 * 2);
  // activations
  float*          x    = (float*)alloc((size_t)B_ * S_ * E_ * 4);
  unsigned short* xb   = (unsigned short*)alloc((size_t)B_ * S_ * E_ * 2);
  unsigned short* xrb  = (unsigned short*)alloc((size_t)B_ * S_ * E_ * 2);
  unsigned short* yb   = (unsigned short*)alloc((size_t)B_ * T_ * E_ * 2);
  float*          xgf  = (float*)alloc((size_t)B_ * S_ * G4 * 4);
  float*          xgb  = (float*)alloc((size_t)B_ * S_ * G4 * 4);
  float*          ypart= (float*)alloc((size_t)B_ * T_ * G4 * 4);
  float*          hs_f = (float*)alloc((size_t)B_ * S_ * H_ * 4);
  float*          hs_b = (float*)alloc((size_t)B_ * S_ * H_ * 4);
  float*          enc  = (float*)alloc((size_t)B_ * S_ * 2 * H_ * 4);
  unsigned short* encb = (unsigned short*)alloc((size_t)B_ * S_ * 2 * H_ * 2);
  float*          encp = (float*)alloc((size_t)B_ * S_ * H_ * 4);
  // states
  float* hf = (float*)alloc((size_t)B_ * H_ * 4);
  float* cf = (float*)alloc((size_t)B_ * H_ * 4);
  float* hb = (float*)alloc((size_t)B_ * H_ * 4);
  float* cbk = (float*)alloc((size_t)B_ * H_ * 4);
  unsigned short* hfb = (unsigned short*)alloc((size_t)B_ * H_ * 2);
  unsigned short* hbb = (unsigned short*)alloc((size_t)B_ * H_ * 2);
  float* gf = (float*)alloc((size_t)B_ * G4 * 4);
  float* gb = (float*)alloc((size_t)B_ * G4 * 4);
  float* c_dec = (float*)alloc((size_t)B_ * H_ * 4);
  float* dech  = (float*)alloc((size_t)B_ * H_ * 4);
  float* gdec  = (float*)alloc((size_t)B_ * G4 * 4);
  unsigned short* A2      = (unsigned short*)alloc((size_t)B_ * 512 * 2);  // [o_prev | h]
  unsigned short* comb_in = (unsigned short*)alloc((size_t)B_ * 768 * 2);  // [dec_h | a_t]
  unsigned short* obuf    = (unsigned short*)alloc((size_t)B_ * H_ * 2);

  auto grid = [](int n, int blk) { return (n + blk - 1) / blk; };

  // ---- weight prep (f32 -> bf16, transposed to N-major) ----
  k_transpose_bf16<<<grid(V_ * H_, 256), 256, 0, stream>>>(vw, V_, vwT, H_, H_, V_);
  k_transpose_bf16<<<grid(G4 * E_, 256), 256, 0, stream>>>(efwx, G4, efwxT, E_, E_, G4);
  k_transpose_bf16<<<grid(G4 * H_, 256), 256, 0, stream>>>(efwh, G4, efwhT, H_, H_, G4);
  k_transpose_bf16<<<grid(G4 * E_, 256), 256, 0, stream>>>(ebwx, G4, ebwxT, E_, E_, G4);
  k_transpose_bf16<<<grid(G4 * H_, 256), 256, 0, stream>>>(ebwh, G4, ebwhT, H_, H_, G4);
  k_transpose_bf16<<<grid(G4 * E_, 256), 256, 0, stream>>>(dwx, G4, dwxTy, E_, E_, G4);
  k_transpose_bf16<<<grid(G4 * H_, 256), 256, 0, stream>>>(dwx + (size_t)E_ * G4, G4, decohT, 512, H_, G4);
  k_transpose_bf16<<<grid(G4 * H_, 256), 256, 0, stream>>>(dwh, G4, decohT + 256, 512, H_, G4);
  k_transpose_bf16<<<grid(H_ * 768, 256), 256, 0, stream>>>(cw, H_, combT, 768, 768, H_);
  k_transpose_bf16<<<grid(H_ * 512, 256), 256, 0, stream>>>(att_proj, H_, attT, 512, 512, H_);

  // ---- embeddings + conv ----
  k_embed_conv<<<grid(B_ * S_ * E_, 256), 256, 0, stream>>>(src_ids, s_emb, conv_k, conv_b, x, xb, xrb);
  k_embed_tgt<<<grid(B_ * T_ * E_, 256), 256, 0, stream>>>(tgt_ids, t_emb, yb);

  // ---- batched input-side GEMMs (WMMA) ----
  // xg_f = x @ enc_fwd_wx + b          (1536 x 1024, K=256)
  k_gemm_bias<<<(96 * 64) / 4, 128, 0, stream>>>(xb, E_, efwxT, E_, efb, xgf, G4, 96, 96 * 64, E_);
  // xg_b = flip(x) @ enc_bwd_wx + b
  k_gemm_bias<<<(96 * 64) / 4, 128, 0, stream>>>(xrb, E_, ebwxT, E_, ebb, xgb, G4, 96, 96 * 64, E_);
  // y_part = y @ dec_wx[:E] + dec_b    (1536 x 1024, K=256)
  k_gemm_bias<<<(96 * 64) / 4, 128, 0, stream>>>(yb, E_, dwxTy, E_, db, ypart, G4, 96, 96 * 64, E_);

  // ---- zero-init recurrent state ----
  hipMemsetAsync(hf, 0, (size_t)B_ * H_ * 4, stream);
  hipMemsetAsync(cf, 0, (size_t)B_ * H_ * 4, stream);
  hipMemsetAsync(hb, 0, (size_t)B_ * H_ * 4, stream);
  hipMemsetAsync(cbk, 0, (size_t)B_ * H_ * 4, stream);
  hipMemsetAsync(hfb, 0, (size_t)B_ * H_ * 2, stream);
  hipMemsetAsync(hbb, 0, (size_t)B_ * H_ * 2, stream);
  hipMemsetAsync(A2, 0, (size_t)B_ * 512 * 2, stream);   // o_prev = 0; h-slot set below

  // ---- encoder recurrence (both directions per step) ----
  for (int s = 0; s < S_; ++s) {
    k_gemm_add_m32<<<16, 128, 0, stream>>>(hfb, H_, efwhT, H_, xgf + s * G4, S_ * G4,
                                           gf, G4, 64, H_);
    k_gemm_add_m32<<<16, 128, 0, stream>>>(hbb, H_, ebwhT, H_, xgb + s * G4, S_ * G4,
                                           gb, G4, 64, H_);
    k_enc_point<<<grid(2 * B_ * H_, 256), 256, 0, stream>>>(gf, gb, cf, hf, hfb, cbk, hb, hbb,
                                                            hs_f, hs_b, src_lens, s);
  }

  // ---- encoder outputs ----
  k_enc_concat<<<grid(B_ * S_ * 2 * H_, 256), 256, 0, stream>>>(hs_f, hs_b, enc, encb);
  k_proj_hc<<<grid(2 * B_ * H_, 256), 256, 0, stream>>>(hf, hb, cf, cbk, h_proj, c_proj, c_dec, A2);
  // enc_proj = enc @ att_proj          (1536 x 256, K=512)
  k_gemm_bias<<<(96 * 16) / 4, 128, 0, stream>>>(encb, 2 * H_, attT, 2 * H_, (const float*)nullptr,
                                                 encp, H_, 96, 96 * 16, 2 * H_);

  // ---- decoder recurrence ----
  for (int t = 0; t < T_; ++t) {
    // g = y_part[t] + [o_prev|h] @ [Wx_opart;Wh]   (32 x 1024, K=512)
    k_gemm_add_m32<<<16, 128, 0, stream>>>(A2, 512, decohT, 512, ypart + t * G4, T_ * G4,
                                           gdec, G4, 64, 512);
    k_dec_point<<<grid(B_ * H_, 256), 256, 0, stream>>>(gdec, c_dec, dech, A2, comb_in);
    k_attn<<<B_, 512, 0, stream>>>(encp, enc, dech, src_lens, comb_in);
    // o_t = tanh([dec_h|a_t] @ comb_w + comb_b)    (32 x 256, K=768) -> obuf & A2 o-slot
    k_gemm_tanh2_m32<<<4, 128, 0, stream>>>(comb_in, 768, combT, 768, cb,
                                            obuf, H_, A2, 512, 16, 768);
    // logits_t = o_t @ vocab_w + vocab_b           (32 x 32000, K=256)
    k_gemm_bias_m32<<<500, 128, 0, stream>>>(obuf, H_, vwT, H_, vb,
                                             out + (long)t * V_, (long)T_ * V_,
                                             2000, H_);
  }
}
